// EncoderLayer_60670708023601
// MI455X (gfx1250) — matrix operand order repaired and checked by reading, not compiled, not run on hardware
//
#include <hip/hip_runtime.h>

typedef _Float16 v16h __attribute__((ext_vector_type(16)));
typedef _Float16 v8h  __attribute__((ext_vector_type(8)));
typedef float    v8f  __attribute__((ext_vector_type(8)));

#define D_MODEL 256
#define D_FC    1024
#define S_TOT   13294
#define NBATCH  2
#define MREAL   (NBATCH * S_TOT)          // 26588
#define MP      26624                     // round up to 64

// ---------------------------------------------------------------------------
// Weight convert: W [K x N] f32 (row-major, as in x @ W) -> Wt [N x K] f16
// ---------------------------------------------------------------------------
__global__ void wconv_kernel(const float* __restrict__ W, _Float16* __restrict__ Wt,
                             int K, int N) {
  int t = blockIdx.x * blockDim.x + threadIdx.x;
  if (t >= N * K) return;
  int n = t / K, k = t - n * K;
  Wt[t] = (_Float16)W[(size_t)k * N + n];
}

// ---------------------------------------------------------------------------
// Activation convert: sh = f16(src), qh = f16(src + pos); zero pad rows
// ---------------------------------------------------------------------------
__global__ void aconv_kernel(const float* __restrict__ src, const float* __restrict__ pos,
                             _Float16* __restrict__ sh, _Float16* __restrict__ qh,
                             int total, int realTotal) {
  int t = blockIdx.x * blockDim.x + threadIdx.x;
  if (t >= total) return;
  float s = 0.f, q = 0.f;
  if (t < realTotal) { s = src[t]; q = s + pos[t]; }
  sh[t] = (_Float16)s;
  qh[t] = (_Float16)q;
}

// ---------------------------------------------------------------------------
// WMMA f16 GEMM:  C[M x N] = A[M x K] * Bt[N x K]^T + bias, optional ReLU,
// output either f32 (Cf) or f16 (Ch). One wave computes a 32x32 tile.
// ---------------------------------------------------------------------------
__device__ __forceinline__ v16h ldfrag(const _Float16* p, int o1, int o2) {
  v8h lo = *(const v8h*)(p + o1);
  v8h hi = *(const v8h*)(p + o2);
  return __builtin_shufflevector(lo, hi, 0,1,2,3,4,5,6,7,8,9,10,11,12,13,14,15);
}

__global__ void __launch_bounds__(256)
gemm_wmma_f16(const _Float16* __restrict__ A, const _Float16* __restrict__ Bt,
              const float* __restrict__ bias,
              float* __restrict__ Cf, _Float16* __restrict__ Ch,
              int M, int N, int K, int relu) {
  const int wave = threadIdx.x >> 5;
  const int lane = threadIdx.x & 31;
  const int half = lane >> 4;
  const int lr   = lane & 15;
  const int mBase = blockIdx.y * 64  + (wave >> 2) * 32;
  const int nBase = blockIdx.x * 128 + (wave & 3) * 32;
  if (mBase >= M || nBase >= N) return;   // wave-uniform: EXEC stays all-1s

  const _Float16* a0p = A  + (size_t)(mBase + lr)      * K;
  const _Float16* a1p = A  + (size_t)(mBase + 16 + lr) * K;
  const _Float16* b0p = Bt + (size_t)(nBase + lr)      * K;
  const _Float16* b1p = Bt + (size_t)(nBase + 16 + lr) * K;

  v8f acc00 = {}, acc01 = {}, acc10 = {}, acc11 = {};
  for (int kk = 0; kk < K; kk += 32) {
    const int ka = kk + half * 8;    // A layout: half-lane K split {0..7|16..23}/{8..15|24..31}
    const int kb = kk + half * 8;    // B layout: half-lane K split {0..7|16..23}/{8..15|24..31}
    v16h a0 = ldfrag(a0p, ka, ka + 16);
    v16h a1 = ldfrag(a1p, ka, ka + 16);
    v16h b0 = ldfrag(b0p, kb, kb + 16);
    v16h b1 = ldfrag(b1p, kb, kb + 16);
    acc00 = __builtin_amdgcn_wmma_f32_16x16x32_f16(false, a0, false, b0, (short)0, acc00, false, false);
    acc01 = __builtin_amdgcn_wmma_f32_16x16x32_f16(false, a0, false, b1, (short)0, acc01, false, false);
    acc10 = __builtin_amdgcn_wmma_f32_16x16x32_f16(false, a1, false, b0, (short)0, acc10, false, false);
    acc11 = __builtin_amdgcn_wmma_f32_16x16x32_f16(false, a1, false, b1, (short)0, acc11, false, false);
  }

  const int col0 = nBase + lr, col1 = col0 + 16;
  const float bs0 = bias[col0], bs1 = bias[col1];
#pragma unroll
  for (int r = 0; r < 8; r++) {
    const int row0 = mBase + half * 8 + r;   // C/D layout: VGPR r -> M = half*8 + r
    const int row1 = row0 + 16;
    float v00 = acc00[r] + bs0, v01 = acc01[r] + bs1;
    float v10 = acc10[r] + bs0, v11 = acc11[r] + bs1;
    if (relu) {
      v00 = fmaxf(v00, 0.f); v01 = fmaxf(v01, 0.f);
      v10 = fmaxf(v10, 0.f); v11 = fmaxf(v11, 0.f);
    }
    if (Ch) {
      Ch[(size_t)row0 * N + col0] = (_Float16)v00;
      Ch[(size_t)row0 * N + col1] = (_Float16)v01;
      Ch[(size_t)row1 * N + col0] = (_Float16)v10;
      Ch[(size_t)row1 * N + col1] = (_Float16)v11;
    } else {
      Cf[(size_t)row0 * N + col0] = v00;
      Cf[(size_t)row0 * N + col1] = v01;
      Cf[(size_t)row1 * N + col0] = v10;
      Cf[(size_t)row1 * N + col1] = v11;
    }
  }
}

// ---------------------------------------------------------------------------
// Softmax over the 16 (level,point) logits per (query, head), in place.
// ---------------------------------------------------------------------------
__global__ void softmax16_kernel(float* __restrict__ attn, int total) {
  int t = blockIdx.x * blockDim.x + threadIdx.x;
  if (t >= total) return;
  float* p = attn + (size_t)t * 16;
  float mx = -3.4e38f;
#pragma unroll
  for (int i = 0; i < 16; i++) mx = fmaxf(mx, p[i]);
  float s = 0.f;
#pragma unroll
  for (int i = 0; i < 16; i++) { float e = expf(p[i] - mx); p[i] = e; s += e; }
  float inv = 1.f / s;
#pragma unroll
  for (int i = 0; i < 16; i++) p[i] *= inv;
}

// ---------------------------------------------------------------------------
// Deformable bilinear sampling. One thread per (query, head); accumulates a
// 32-dim head vector over 4 levels x 4 points x 4 taps. f16 output for GEMM.
// ---------------------------------------------------------------------------
__global__ void deform_sample_kernel(const float* __restrict__ value,
                                     const float* __restrict__ off,
                                     const float* __restrict__ attn,
                                     const float* __restrict__ refp,
                                     _Float16* __restrict__ out,
                                     int totalMH, int Mreal) {
  int t = blockIdx.x * blockDim.x + threadIdx.x;
  if (t >= totalMH) return;
  const int m = t >> 3;
  const int h = t & 7;

  float4 acc[8] = {};
  if (m < Mreal) {
    const int Hs[4] = {100, 50, 25, 13};
    const int Ws[4] = {100, 50, 25, 13};
    const int St[4] = {0, 10000, 12500, 13125};
    const int b = m / S_TOT;
    const float* offrow = off  + (size_t)m * 256 + h * 32;
    const float* attrow = attn + (size_t)m * 128 + h * 16;
    const float* refrow = refp + (size_t)m * 8;
    const int vbase = b * S_TOT;
#pragma unroll
    for (int l = 0; l < 4; l++) {
      const float Wl = (float)Ws[l], Hl = (float)Hs[l];
      const float rx = refrow[l * 2 + 0], ry = refrow[l * 2 + 1];
      const float* vlev = value + ((size_t)(vbase + St[l])) * 256 + h * 32;
#pragma unroll
      for (int p = 0; p < 4; p++) {
        const float px = rx * Wl + offrow[l * 8 + p * 2 + 0] - 0.5f;
        const float py = ry * Hl + offrow[l * 8 + p * 2 + 1] - 0.5f;
        const float x0f = floorf(px), y0f = floorf(py);
        const float wx = px - x0f, wy = py - y0f;
        const int x0 = (int)x0f, y0 = (int)y0f;
        const float aw = attrow[l * 4 + p];
        const float tw[4] = { (1.f - wx) * (1.f - wy) * aw, wx * (1.f - wy) * aw,
                              (1.f - wx) * wy * aw,         wx * wy * aw };
        const int tx[4] = {x0, x0 + 1, x0, x0 + 1};
        const int ty[4] = {y0, y0, y0 + 1, y0 + 1};
#pragma unroll
        for (int tp = 0; tp < 4; tp++) {
          const int xi = tx[tp], yi = ty[tp];
          if (xi < 0 || xi >= Ws[l] || yi < 0 || yi >= Hs[l]) continue;
          const float w = tw[tp];
          const float4* vp = (const float4*)(vlev + ((size_t)yi * Ws[l] + xi) * 256);
#pragma unroll
          for (int j = 0; j < 8; j++) {
            float4 v = vp[j];
            acc[j].x += w * v.x; acc[j].y += w * v.y;
            acc[j].z += w * v.z; acc[j].w += w * v.w;
          }
        }
      }
    }
  }
  _Float16* orow = out + (size_t)m * 256 + h * 32;
#pragma unroll
  for (int j = 0; j < 8; j++) {
    orow[j * 4 + 0] = (_Float16)acc[j].x;
    orow[j * 4 + 1] = (_Float16)acc[j].y;
    orow[j * 4 + 2] = (_Float16)acc[j].z;
    orow[j * 4 + 3] = (_Float16)acc[j].w;
  }
}

// ---------------------------------------------------------------------------
// Fused residual + LayerNorm: y = LN(a + b)*g + beta. One wave32 per row.
// Optionally also writes an f16 copy (zero-padded past Mreal) for next GEMM.
// ---------------------------------------------------------------------------
__global__ void ln_fused_kernel(const float* __restrict__ A, const float* __restrict__ Bb,
                                const float* __restrict__ g, const float* __restrict__ beta,
                                float* __restrict__ outf, _Float16* __restrict__ outh,
                                int Mrows, int Mreal) {
  const int row  = blockIdx.x * (blockDim.x >> 5) + (threadIdx.x >> 5);
  const int lane = threadIdx.x & 31;
  if (row >= Mrows) return;
  if (row >= Mreal) {
    if (outh) {
#pragma unroll
      for (int j = 0; j < 8; j++) outh[(size_t)row * 256 + j * 32 + lane] = (_Float16)0.f;
    }
    return;
  }
  const float* ra = A  + (size_t)row * 256;
  const float* rb = Bb + (size_t)row * 256;
  float x[8];
  float s = 0.f;
#pragma unroll
  for (int j = 0; j < 8; j++) { x[j] = ra[j * 32 + lane] + rb[j * 32 + lane]; s += x[j]; }
  for (int o = 16; o > 0; o >>= 1) s += __shfl_xor(s, o, 32);
  const float mu = s * (1.f / 256.f);
  float v = 0.f;
#pragma unroll
  for (int j = 0; j < 8; j++) { float d = x[j] - mu; v += d * d; }
  for (int o = 16; o > 0; o >>= 1) v += __shfl_xor(v, o, 32);
  const float inv = rsqrtf(v * (1.f / 256.f) + 1e-5f);
#pragma unroll
  for (int j = 0; j < 8; j++) {
    const int c = j * 32 + lane;
    const float y = (x[j] - mu) * inv * g[c] + beta[c];
    if (outf) outf[(size_t)row * 256 + c] = y;
    if (outh) outh[(size_t)row * 256 + c] = (_Float16)y;
  }
}

// ---------------------------------------------------------------------------
extern "C" void kernel_launch(void* const* d_in, const int* in_sizes, int n_in,
                              void* d_out, int out_size, void* d_ws, size_t ws_size,
                              hipStream_t stream) {
  (void)in_sizes; (void)n_in; (void)out_size; (void)ws_size;
  const float* src   = (const float*)d_in[0];
  const float* pos   = (const float*)d_in[1];
  const float* refp  = (const float*)d_in[2];
  // d_in[3] spatial_shapes, d_in[4] level_start_index: compile-time constants
  const float* W_off  = (const float*)d_in[5];
  const float* b_off  = (const float*)d_in[6];
  const float* W_attn = (const float*)d_in[7];
  const float* b_attn = (const float*)d_in[8];
  const float* W_val  = (const float*)d_in[9];
  const float* b_val  = (const float*)d_in[10];
  const float* W_out  = (const float*)d_in[11];
  const float* b_out  = (const float*)d_in[12];
  const float* ln1_g  = (const float*)d_in[13];
  const float* ln1_b  = (const float*)d_in[14];
  const float* W1     = (const float*)d_in[15];
  const float* b1     = (const float*)d_in[16];
  const float* W2     = (const float*)d_in[17];
  const float* b2     = (const float*)d_in[18];
  const float* ln2_g  = (const float*)d_in[19];
  const float* ln2_b  = (const float*)d_in[20];
  float* out = (float*)d_out;

  // -------- workspace layout (deterministic) --------
  char* ws = (char*)d_ws;
  size_t o = 0;
  auto take = [&](size_t bytes) -> char* {
    char* p = ws + o;
    o = (o + bytes + 255) & ~(size_t)255;
    return p;
  };
  float*    VALUE = (float*)   take((size_t)MP * 256 * 4);
  float*    OFFB  = (float*)   take((size_t)MP * 256 * 4);
  float*    ATTN  = (float*)   take((size_t)MP * 128 * 4);
  _Float16* QH    = (_Float16*)take((size_t)MP * 256 * 2);
  _Float16* SH    = (_Float16*)take((size_t)MP * 256 * 2);
  _Float16* HH    = (_Float16*)take((size_t)MP * 1024 * 2);
  _Float16* WvT   = (_Float16*)take(65536 * 2);
  _Float16* WoT   = (_Float16*)take(65536 * 2);
  _Float16* WaT   = (_Float16*)take(32768 * 2);
  _Float16* WuT   = (_Float16*)take(65536 * 2);
  _Float16* W1T   = (_Float16*)take(262144 * 2);
  _Float16* W2T   = (_Float16*)take(262144 * 2);
  // buffer reuse (dataflow-safe under stream ordering):
  _Float16* SAMPH   = SH;     // src f16 dead after value GEMM
  float*    ATTNOUT = VALUE;  // value dead after sampling
  float*    X       = OFFB;   // offsets dead after sampling
  _Float16* XH      = QH;     // q f16 dead after off/attn GEMMs
  float*    FFN     = VALUE;  // attn_out dead after LN1

  // -------- weights -> f16 transposed [N x K] --------
  wconv_kernel<<<(65536  + 255) / 256, 256, 0, stream>>>(W_val,  WvT, 256,  256);
  wconv_kernel<<<(65536  + 255) / 256, 256, 0, stream>>>(W_off,  WoT, 256,  256);
  wconv_kernel<<<(32768  + 255) / 256, 256, 0, stream>>>(W_attn, WaT, 256,  128);
  wconv_kernel<<<(65536  + 255) / 256, 256, 0, stream>>>(W_out,  WuT, 256,  256);
  wconv_kernel<<<(262144 + 255) / 256, 256, 0, stream>>>(W1,     W1T, 256,  1024);
  wconv_kernel<<<(262144 + 255) / 256, 256, 0, stream>>>(W2,     W2T, 1024, 256);

  // -------- activations -> f16 (src, src+pos), zero-padded --------
  aconv_kernel<<<(MP * 256) / 256, 256, 0, stream>>>(src, pos, SH, QH,
                                                     MP * 256, MREAL * 256);

  const dim3 blk(256);
  const dim3 gN256(256 / 128, MP / 64);
  const dim3 gN128(1, MP / 64);
  const dim3 gN1024(1024 / 128, MP / 64);

  // -------- projections (WMMA) --------
  gemm_wmma_f16<<<gN256, blk, 0, stream>>>(SH, WvT, b_val,  VALUE, nullptr, MP, 256, 256, 0);
  gemm_wmma_f16<<<gN256, blk, 0, stream>>>(QH, WoT, b_off,  OFFB,  nullptr, MP, 256, 256, 0);
  gemm_wmma_f16<<<gN128, blk, 0, stream>>>(QH, WaT, b_attn, ATTN,  nullptr, MP, 128, 256, 0);

  // -------- softmax + deformable sampling --------
  softmax16_kernel<<<(MREAL * 8 + 255) / 256, 256, 0, stream>>>(ATTN, MREAL * 8);
  deform_sample_kernel<<<(MP * 8 + 255) / 256, 256, 0, stream>>>(VALUE, OFFB, ATTN, refp,
                                                                 SAMPH, MP * 8, MREAL);

  // -------- output projection + LN1 --------
  gemm_wmma_f16<<<gN256, blk, 0, stream>>>(SAMPH, WuT, b_out, ATTNOUT, nullptr, MP, 256, 256, 0);
  ln_fused_kernel<<<MP / 8, 256, 0, stream>>>(src, ATTNOUT, ln1_g, ln1_b, X, XH, MP, MREAL);

  // -------- FFN (ReLU fused into GEMM1, f16 hidden) + LN2 --------
  gemm_wmma_f16<<<gN1024, blk, 0, stream>>>(XH, W1T, b1, nullptr, HH, MP, 1024, 256, 1);
  gemm_wmma_f16<<<gN256,  blk, 0, stream>>>(HH, W2T, b2, FFN, nullptr, MP, 256, 1024, 0);
  ln_fused_kernel<<<(MREAL + 7) / 8, 256, 0, stream>>>(X, FFN, ln2_g, ln2_b,
                                                       out, nullptr, MREAL, MREAL);
}
